// ACTP_6047313953604
// MI455X (gfx1250) — matrix operand order repaired and checked by reading, not compiled, not run on hardware
//
#include <hip/hip_runtime.h>
#include <hip/hip_bf16.h>

// ---------------- types ----------------
typedef __attribute__((ext_vector_type(16))) __bf16       v16bf;
typedef __attribute__((ext_vector_type(8)))  float        v8f;
typedef __attribute__((ext_vector_type(4)))  unsigned int v4u;

union FragU { v16bf v; v4u q[2]; };

#define H200   200
#define NTG    52      // gate n-tiles: 13 slices * 4 gates (208-padded hidden)
#define KT_X   2       // x:   K=48  -> 64
#define KT_H   7       // h:   K=200 -> 224
#define KT_A   8       // a_t: K=248 -> 256
#define NT_FC1 13
#define KT_FC1 8
#define NT_FC2 3
#define KT_FC2 7
#define TSTEPS 511
#define BATCH  512

__device__ inline unsigned short f2bf(float f) {
  unsigned int u = __float_as_uint(f);
  unsigned int r = u + 0x7FFFu + ((u >> 16) & 1u);
  return (unsigned short)(r >> 16);
}

__device__ inline float fsig(float x)  { return 1.0f / (1.0f + __expf(-x)); }
__device__ inline float ftanh(float x) {
  x = fminf(fmaxf(x, -15.0f), 15.0f);
  float e = __expf(-2.0f * x);
  return (1.0f - e) / (1.0f + e);
}

// A fragment (16x32 bf16, M=batch) from row-major bf16 LDS (rowStride in elems, mult of 8).
__device__ inline v16bf loadA(const unsigned short* sh, int rowStride, int kt, int lane) {
  int m  = lane & 15;
  int hi = (lane >> 4) << 3;                       // 0 or 8
  const unsigned short* p = sh + m * rowStride + kt * 32 + hi;
  FragU f;
  f.q[0] = *(const v4u*)(p);                       // K = base+0..7
  f.q[1] = *(const v4u*)(p + 16);                  // K = base+16..23
  return f.v;
}

// B fragment (32x16 bf16) pre-packed fragment-major in global: lane*16 contiguous.
__device__ inline v16bf loadB(const unsigned short* gp, int lane) {
  const unsigned short* p = gp + lane * 16;
  FragU f;
  f.q[0] = *(const v4u*)(p);
  f.q[1] = *(const v4u*)(p + 8);
  return f.v;
}

__device__ inline v8f wmma_bf(v16bf a, v16bf b, v8f c) {
  return __builtin_amdgcn_wmma_f32_16x16x32_bf16(false, a, false, b, (short)0, c, false, false);
}

// One hidden slice's gate GEMM, fully unrolled K chain. The slice-validity
// guard lives OUTSIDE this function (one scalar branch per phase, none per
// K-iteration -> no per-kt phi copies). Per K-tile: 4 B-fragment loads are
// issued before the WMMAs, so 8 global_load_b128 stay in flight.
template <int KT>
__device__ inline void gemm_slice(const unsigned short* __restrict__ Bbase,
                                  const unsigned short* sh, int rowStride,
                                  int s, int lane, v8f (&acc4)[4]) {
#pragma unroll
  for (int kt = 0; kt < KT; ++kt) {
    v16bf bb[4];
#pragma unroll
    for (int g = 0; g < 4; ++g)
      bb[g] = loadB(Bbase + ((size_t)((4 * s + g) * KT + kt) << 9), lane);
    v16bf a = loadA(sh, rowStride, kt, lane);
#pragma unroll
    for (int g = 0; g < 4; ++g)
      acc4[g] = wmma_bf(a, bb[g], acc4[g]);
  }
}

// LSTM epilogue: gates in registers -> c (register), h -> two LDS buffers.
__device__ inline void lstm_ep(const v8f* acc4, v8f& c, const float* bias, int s, int lane,
                               unsigned short* d0, int st0, unsigned short* d1, int st1) {
  int nl   = lane & 15;
  int mh   = (lane >> 4) << 3;
  int base = 64 * s + nl;
  float bi = bias[base], bf = bias[base + 16], bg = bias[base + 32], bo = bias[base + 48];
  int hidx = s * 16 + nl;
#pragma unroll
  for (int r = 0; r < 8; ++r) {
    float gi = fsig (acc4[0][r] + bi);
    float gf = fsig (acc4[1][r] + bf);
    float gg = ftanh(acc4[2][r] + bg);
    float go = fsig (acc4[3][r] + bo);
    float cn = gf * c[r] + gi * gg;
    c[r] = cn;
    float hn = go * ftanh(cn);
    if (hidx < H200) {
      unsigned short hb = f2bf(hn);
      int m = r + mh;
      d0[m * st0 + hidx] = hb;
      d1[m * st1 + hidx] = hb;
    }
  }
}

// -------- prep: f32 weights -> bf16 WMMA B-fragments (optionally gate-interleaved) --------
__global__ void prep_w(const float* __restrict__ W, unsigned short* __restrict__ dst,
                       int NT, int KT, int rowsOrig, int colsOrig, int gateReorder) {
  int idx   = blockIdx.x * 256 + threadIdx.x;
  int total = NT * KT * 512;
  if (idx >= total) return;
  int tile = idx >> 9;
  int e    = idx & 511;
  int lane = e >> 4;
  int j    = e & 15;
  int nt   = tile / KT;
  int kt   = tile - nt * KT;
  int nl   = lane & 15;
  int k    = kt * 32 + ((lane >> 4) << 3) + (j < 8 ? j : j + 8);   // lanes 0-15: K 0..7,16..23 ; 16-31: K 8..15,24..31
  int r, valid;
  if (gateReorder) {                              // ntile = 4*slice + gate
    int g = nt & 3, s = nt >> 2, h = s * 16 + nl;
    r = g * H200 + h;  valid = (h < H200);
  } else {
    r = nt * 16 + nl;  valid = (r < rowsOrig);
  }
  float v = (valid && k < colsOrig) ? W[r * colsOrig + k] : 0.0f;
  dst[idx] = f2bf(v);
}

__global__ void prep_bias(const float* __restrict__ bih, const float* __restrict__ bhh,
                          float* __restrict__ dst) {
  int j = blockIdx.x * 256 + threadIdx.x;
  if (j >= NTG * 16) return;
  int nt = j >> 4, nl = j & 15, g = nt & 3, s = nt >> 2, h = s * 16 + nl;
  dst[j] = (h < H200) ? (bih[g * H200 + h] + bhh[g * H200 + h]) : 0.0f;
}

// ---------------- persistent recurrent kernel: 1 WG = 16 batch rows, all 511 steps ----------------
__global__ __launch_bounds__(256) void actp_main(
    const float* __restrict__ tactiles, const float* __restrict__ actions,
    const float* __restrict__ fc0_b, const float* __restrict__ fc1_b, const float* __restrict__ fc2_b,
    const unsigned short* __restrict__ B1ih, const unsigned short* __restrict__ B1hh,
    const unsigned short* __restrict__ B2ih, const unsigned short* __restrict__ B2hh,
    const unsigned short* __restrict__ Bfc0, const unsigned short* __restrict__ Bfc1,
    const unsigned short* __restrict__ Bfc2,
    const float* __restrict__ bias1, const float* __restrict__ bias2,
    const int* __restrict__ cfp, float* __restrict__ out) {
  __shared__ __attribute__((aligned(16))) unsigned short xbuf [16 * 64];   // x, K-padded
  __shared__ __attribute__((aligned(16))) unsigned short h1buf[16 * 224];  // h1, K-padded
  __shared__ __attribute__((aligned(16))) unsigned short h2buf[16 * 224];  // h2, K-padded
  __shared__ __attribute__((aligned(16))) unsigned short a_t  [16 * 256];  // [h1 | out0 | 0]
  __shared__ __attribute__((aligned(16))) unsigned short lp   [16 * 256];  // [h2 | x    | 0]
  __shared__ __attribute__((aligned(16))) unsigned short out3b[16 * 224];  // out3, K-padded
  __shared__ __attribute__((aligned(16))) unsigned short tiled[16 * 32];   // fc0 input
  __shared__ float out4l[16 * 48];

  const int tid  = threadIdx.x;
  const int lane = tid & 31;
  // wave index as a compiler-visible scalar: uniform guards become s_cbranch,
  // EXEC stays all-ones through every WMMA (ISA 7.12 requirement), and all
  // B-panel base addresses become SGPR arithmetic.
  const int wave = __builtin_amdgcn_readfirstlane(tid >> 5);
  const int b0   = blockIdx.x * 16;
  const int CF   = cfp[0];

  for (int i = tid; i < 16 * 64;  i += 256) xbuf[i]  = 0;
  for (int i = tid; i < 16 * 224; i += 256) h1buf[i] = 0;
  for (int i = tid; i < 16 * 224; i += 256) h2buf[i] = 0;
  for (int i = tid; i < 16 * 256; i += 256) a_t[i]   = 0;
  for (int i = tid; i < 16 * 256; i += 256) lp[i]    = 0;
  for (int i = tid; i < 16 * 224; i += 256) out3b[i] = 0;
  for (int i = tid; i < 16 * 32;  i += 256) tiled[i] = 0;
  for (int i = tid; i < 16 * 48;  i += 256) out4l[i] = 0.0f;
  __syncthreads();

  v8f c1s[2], c2s[2];
#pragma unroll
  for (int i = 0; i < 2; ++i) { c1s[i] = (v8f){0,0,0,0,0,0,0,0}; c2s[i] = (v8f){0,0,0,0,0,0,0,0}; }

  for (int t = 0; t < TSTEPS; ++t) {
    // ---- phase 1: build x (tactile or fed-back out4) and fc0 input ----
    for (int i = tid; i < 16 * 48; i += 256) {
      int m = i / 48, c = i % 48;
      float v = (t < CF) ? tactiles[((size_t)t * BATCH + b0 + m) * 48 + c] : out4l[i];
      unsigned short bv = f2bf(v);
      xbuf[m * 64 + c]        = bv;
      lp  [m * 256 + 200 + c] = bv;
    }
    for (int i = tid; i < 16 * 24; i += 256) {
      int m = i / 24, c = i % 24;
      int src_t = (c < 12) ? 0 : (t + 1);            // [state,state,act,act]
      float v = actions[((size_t)src_t * BATCH + b0 + m) * 6 + (c % 6)];
      tiled[m * 32 + c] = f2bf(v);
    }
    __syncthreads();

    // ---- phase 2: LSTM1 GEMM (+ fc0 GEMM) ----
    v8f acc[2][4];
#pragma unroll
    for (int i = 0; i < 2; ++i)
#pragma unroll
      for (int g = 0; g < 4; ++g) acc[i][g] = (v8f){0,0,0,0,0,0,0,0};
#pragma unroll
    for (int i = 0; i < 2; ++i) {
      int s = wave + i * 8;
      if (s < 13) {                      // one scalar branch per slice per phase
        gemm_slice<KT_X>(B1ih, xbuf,  64,  s, lane, acc[i]);
        gemm_slice<KT_H>(B1hh, h1buf, 224, s, lane, acc[i]);
      }
    }
    v8f acc0 = (v8f){0,0,0,0,0,0,0,0};
    if (wave < 3) {
      v16bf b = loadB(Bfc0 + ((size_t)wave << 9), lane);
      v16bf a = loadA(tiled, 32, 0, lane);
      acc0 = wmma_bf(a, b, acc0);
    }
    __syncthreads();   // all h1buf reads done before overwrite

    // ---- phase 3: LSTM1 epilogue (c1 stays in VGPRs) + fc0 relu ----
#pragma unroll
    for (int i = 0; i < 2; ++i) {
      int s = wave + i * 8;
      if (s < 13) lstm_ep(acc[i], c1s[i], bias1, s, lane, h1buf, 224, a_t, 256);
    }
    if (wave < 3) {
      int nl = lane & 15, col = wave * 16 + nl, mh = (lane >> 4) << 3;
      float bb = fc0_b[col];
#pragma unroll
      for (int r = 0; r < 8; ++r) {
        float v = fmaxf(acc0[r] + bb, 0.0f);
        a_t[(r + mh) * 256 + 200 + col] = f2bf(v);
      }
    }
    __syncthreads();

    // ---- phase 4: LSTM2 GEMM ----
#pragma unroll
    for (int i = 0; i < 2; ++i)
#pragma unroll
      for (int g = 0; g < 4; ++g) acc[i][g] = (v8f){0,0,0,0,0,0,0,0};
#pragma unroll
    for (int i = 0; i < 2; ++i) {
      int s = wave + i * 8;
      if (s < 13) {
        gemm_slice<KT_A>(B2ih, a_t,   256, s, lane, acc[i]);
        gemm_slice<KT_H>(B2hh, h2buf, 224, s, lane, acc[i]);
      }
    }
    __syncthreads();

    // ---- phase 5: LSTM2 epilogue ----
#pragma unroll
    for (int i = 0; i < 2; ++i) {
      int s = wave + i * 8;
      if (s < 13) lstm_ep(acc[i], c2s[i], bias2, s, lane, h2buf, 224, lp, 256);
    }
    __syncthreads();

    // ---- phase 6: fc1 = tanh(lp @ fc1_w^T + b) ----
    // Accumulators are dead here: per N-tile (guard outside the K loop),
    // prefetch the whole 8-fragment B panel, then run the unrolled WMMA chain.
#pragma unroll
    for (int i = 0; i < 2; ++i) {
      int nt = wave + i * 8;
      if (nt < NT_FC1) {
        v16bf bf1[KT_FC1];
#pragma unroll
        for (int kt = 0; kt < KT_FC1; ++kt)
          bf1[kt] = loadB(Bfc1 + ((size_t)(nt * KT_FC1 + kt) << 9), lane);
        v8f accf = (v8f){0,0,0,0,0,0,0,0};
#pragma unroll
        for (int kt = 0; kt < KT_FC1; ++kt)
          accf = wmma_bf(loadA(lp, 256, kt, lane), bf1[kt], accf);
        int nl = lane & 15, col = nt * 16 + nl, mh = (lane >> 4) << 3;
        if (col < H200) {
          float bb = fc1_b[col];
#pragma unroll
          for (int r = 0; r < 8; ++r)
            out3b[(r + mh) * 224 + col] = f2bf(ftanh(accf[r] + bb));
        }
      }
    }
    __syncthreads();

    // ---- phase 7: fc2 = tanh(out3 @ fc2_w^T + b) -> out4 (LDS + global) ----
    if (wave < 3) {
      v16bf bf2[KT_FC2];
#pragma unroll
      for (int kt = 0; kt < KT_FC2; ++kt)
        bf2[kt] = loadB(Bfc2 + ((size_t)(wave * KT_FC2 + kt) << 9), lane);
      v8f acc2 = (v8f){0,0,0,0,0,0,0,0};
#pragma unroll
      for (int kt = 0; kt < KT_FC2; ++kt)
        acc2 = wmma_bf(loadA(out3b, 224, kt, lane), bf2[kt], acc2);
      int nl = lane & 15, col = wave * 16 + nl, mh = (lane >> 4) << 3;
      float bb = fc2_b[col];
#pragma unroll
      for (int r = 0; r < 8; ++r) {
        int m = r + mh;
        float v = ftanh(acc2[r] + bb);
        out4l[m * 48 + col] = v;
        if (t >= CF - 1)
          out[((size_t)(t - (CF - 1)) * BATCH + b0 + m) * 48 + col] = v;
      }
    }
    __syncthreads();   // protect out4l / lp / xbuf before next step
  }
}

// ---------------- host side ----------------
extern "C" void kernel_launch(void* const* d_in, const int* in_sizes, int n_in,
                              void* d_out, int out_size, void* d_ws, size_t ws_size,
                              hipStream_t stream) {
  (void)in_sizes; (void)n_in; (void)out_size; (void)ws_size;
  const float* tactiles = (const float*)d_in[0];
  const float* actions  = (const float*)d_in[1];
  const float* fc0_w    = (const float*)d_in[2];
  const float* fc0_b    = (const float*)d_in[3];
  const float* l1_wih   = (const float*)d_in[4];
  const float* l1_whh   = (const float*)d_in[5];
  const float* l1_bih   = (const float*)d_in[6];
  const float* l1_bhh   = (const float*)d_in[7];
  const float* l2_wih   = (const float*)d_in[8];
  const float* l2_whh   = (const float*)d_in[9];
  const float* l2_bih   = (const float*)d_in[10];
  const float* l2_bhh   = (const float*)d_in[11];
  const float* fc1_w    = (const float*)d_in[12];
  const float* fc1_b    = (const float*)d_in[13];
  const float* fc2_w    = (const float*)d_in[14];
  const float* fc2_b    = (const float*)d_in[15];
  const int*   cfp      = (const int*)d_in[16];

  unsigned short* ws = (unsigned short*)d_ws;
  size_t o = 0;
  unsigned short* B1ih = ws + o; o += (size_t)NTG * KT_X   * 512;
  unsigned short* B1hh = ws + o; o += (size_t)NTG * KT_H   * 512;
  unsigned short* B2ih = ws + o; o += (size_t)NTG * KT_A   * 512;
  unsigned short* B2hh = ws + o; o += (size_t)NTG * KT_H   * 512;
  unsigned short* Bfc0 = ws + o; o += (size_t)3   * 1      * 512;
  unsigned short* Bfc1 = ws + o; o += (size_t)NT_FC1 * KT_FC1 * 512;
  unsigned short* Bfc2 = ws + o; o += (size_t)NT_FC2 * KT_FC2 * 512;
  float* bias1 = (float*)(ws + o);
  float* bias2 = bias1 + NTG * 16;

  auto blocks = [](int n) { return (n + 255) / 256; };
  prep_w<<<blocks(NTG * KT_X   * 512), 256, 0, stream>>>(l1_wih, B1ih, NTG, KT_X, 800, 48, 1);
  prep_w<<<blocks(NTG * KT_H   * 512), 256, 0, stream>>>(l1_whh, B1hh, NTG, KT_H, 800, 200, 1);
  prep_w<<<blocks(NTG * KT_A   * 512), 256, 0, stream>>>(l2_wih, B2ih, NTG, KT_A, 800, 248, 1);
  prep_w<<<blocks(NTG * KT_H   * 512), 256, 0, stream>>>(l2_whh, B2hh, NTG, KT_H, 800, 200, 1);
  prep_w<<<blocks(3 * 512),            256, 0, stream>>>(fc0_w,  Bfc0, 3, 1, 48, 24, 0);
  prep_w<<<blocks(NT_FC1*KT_FC1*512),  256, 0, stream>>>(fc1_w,  Bfc1, NT_FC1, KT_FC1, 200, 248, 0);
  prep_w<<<blocks(NT_FC2*KT_FC2*512),  256, 0, stream>>>(fc2_w,  Bfc2, NT_FC2, KT_FC2, 48, 200, 0);
  prep_bias<<<blocks(NTG * 16), 256, 0, stream>>>(l1_bih, l1_bhh, bias1);
  prep_bias<<<blocks(NTG * 16), 256, 0, stream>>>(l2_bih, l2_bhh, bias2);

  actp_main<<<32, 256, 0, stream>>>(tactiles, actions, fc0_b, fc1_b, fc2_b,
                                    B1ih, B1hh, B2ih, B2hh, Bfc0, Bfc1, Bfc2,
                                    bias1, bias2, cfp, (float*)d_out);
}
